// SynthesisBlock_16192026706285
// MI455X (gfx1250) — matrix operand order repaired and checked by reading, not compiled, not run on hardware
//
#include <hip/hip_runtime.h>
#include <hip/hip_bf16.h>

// ---------------------------------------------------------------------------
// SynthesisBlock for MI455X (gfx1250, wave32, WMMA f16 -> f32 accumulate)
// Activations stored NHWC (channel-innermost) so WMMA A/B fragments are
// direct aligned b128 global loads -- no LDS staging, no predication.
// GEMMs register-blocked 4(M-tiles) x 2(N-tiles): 12 b128 loads per 8 wmma.
// ---------------------------------------------------------------------------

typedef _Float16 half_t;
typedef __attribute__((ext_vector_type(16))) _Float16 v16h;
typedef __attribute__((ext_vector_type(8)))  float    v8f;

static constexpr int   CIN_  = 512;
static constexpr int   COUT_ = 512;
static constexpr float SQ2_  = 1.41421356237309504880f;   // relu gain
static constexpr float SQH_  = 0.70710678118654752440f;   // residual gain
static constexpr float GAIN512_ = 0.04419417382415922f;   // 1/sqrt(512)

union Frag  { v16h h; unsigned u[8]; };
union Pack8 { uint4 q; half_t h[8]; };

// A fragment (16x32 f16, ISA 7.12.2): lane(g,m) holds K halves [8g,8g+8) and
// [16+8g,16+8g+8) of row m. row0 = first row of the 16-row tile, row stride 512.
__device__ __forceinline__ void load_frag_a(Frag& f, const half_t* __restrict__ row0,
                                            unsigned g, unsigned mn)
{
  const half_t* p = row0 + (long)mn * 512 + 8 * (int)g;
  uint4 lo = *reinterpret_cast<const uint4*>(p);
  uint4 hi = *reinterpret_cast<const uint4*>(p + 16);
  f.u[0] = lo.x; f.u[1] = lo.y; f.u[2] = lo.z; f.u[3] = lo.w;
  f.u[4] = hi.x; f.u[5] = hi.y; f.u[6] = hi.z; f.u[7] = hi.w;
}

// B fragment (32x16 f16): lane(g,n) holds K halves [8g,8g+8) and [16+8g,16+8g+8) of position n.
// pos0 = channel pointer of position 0 (NHWC: position stride = 512 halves).
__device__ __forceinline__ void load_frag_b(Frag& f, const half_t* __restrict__ pos0,
                                            unsigned g, unsigned mn)
{
  const half_t* p = pos0 + (long)mn * 512 + 8 * (int)g;
  uint4 lo = *reinterpret_cast<const uint4*>(p);
  uint4 hi = *reinterpret_cast<const uint4*>(p + 16);
  f.u[0] = lo.x; f.u[1] = lo.y; f.u[2] = lo.z; f.u[3] = lo.w;
  f.u[4] = hi.x; f.u[5] = hi.y; f.u[6] = hi.z; f.u[7] = hi.w;
}

__device__ __forceinline__ v8f wmma_step(const Frag& a, const Frag& b, v8f c)
{
  return __builtin_amdgcn_wmma_f32_16x16x32_f16(false, a.h, false, b.h,
                                                (short)0, c, false, false);
}

// ---------------------------------------------------------------------------
// Small prep kernels (latency-bound, negligible vs GEMMs)
// ---------------------------------------------------------------------------

// style s[b,j] = (1/sqrt(512)) * dot(latent_row[b], aff_w[j]) + aff_b[j]
__global__ void k_affine(const float* __restrict__ lat, long latBStride,
                         const float* __restrict__ w, const float* __restrict__ bias,
                         float* __restrict__ s)
{
  int idx = blockIdx.x * blockDim.x + threadIdx.x;      // 16*512
  int j = idx & 511, b = idx >> 9;
  const float* lr = lat + (long)b * latBStride;
  const float* wr = w + (long)j * 512;
  float acc = 0.f;
  for (int l = 0; l < 512; ++l) acc += lr[l] * wr[l];
  s[idx] = acc * GAIN512_ + bias[j];
}

// wsum[o,i] = sum_{3x3} w^2
__global__ void k_wsum(const float* __restrict__ w, float* __restrict__ ws)
{
  int idx = blockIdx.x * blockDim.x + threadIdx.x;      // 512*512
  const float* p = w + (long)idx * 9;
  float a = 0.f;
#pragma unroll
  for (int k = 0; k < 9; ++k) a += p[k] * p[k];
  ws[idx] = a;
}

// d[b,o] = rsqrt(sum_i s[b,i]^2 * wsum[o,i] + 1e-8)
__global__ void k_demod(const float* __restrict__ s, const float* __restrict__ wsum,
                        float* __restrict__ d)
{
  int idx = blockIdx.x * blockDim.x + threadIdx.x;      // 16*512
  int o = idx & 511, b = idx >> 9;
  const float* sr = s + (long)b * 512;
  const float* wr = wsum + (long)o * 512;
  float a = 0.f;
  for (int i = 0; i < 512; ++i) a += sr[i] * sr[i] * wr[i];
  d[idx] = rsqrtf(a + 1e-8f);
}

// wT[t][o][i] = (f16) w[o][i][ flip ? 8-t : t ]   (flip -> conv_transpose taps)
__global__ void k_wprep(const float* __restrict__ w, half_t* __restrict__ wt, int flip)
{
  long idx = (long)blockIdx.x * blockDim.x + threadIdx.x;   // 9*512*512
  int t = (int)(idx >> 18);
  long oi = idx & 262143;
  int k = flip ? (8 - t) : t;
  wt[idx] = (half_t)w[oi * 9 + k];
}

// skip weight: fold 1/sqrt(CIN) gain, convert f16
__global__ void k_wskp(const float* __restrict__ w, half_t* __restrict__ wt)
{
  int idx = blockIdx.x * blockDim.x + threadIdx.x;          // 512*512
  wt[idx] = (half_t)(w[idx] * GAIN512_);
}

// modulated conv0 input, zero-padded, NHWC: xm[b][1+y][1+x][c] = x * s0[b,c]
__global__ void k_xmod0(const float* __restrict__ x, const float* __restrict__ s0,
                        half_t* __restrict__ xm)
{
  long idx = (long)blockIdx.x * blockDim.x + threadIdx.x;   // 16*34*34*512
  int c = (int)(idx & 511);
  long t = idx >> 9;
  int xx = (int)(t % 34); t /= 34;
  int yy = (int)(t % 34); int b = (int)(t / 34);
  half_t v = (half_t)0.0f;
  if (xx >= 1 && xx <= 32 && yy >= 1 && yy <= 32)
    v = (half_t)(x[(((long)(b * 512 + c)) * 32 + (yy - 1)) * 32 + (xx - 1)]
                 * s0[b * 512 + c]);
  xm[idx] = v;
}

// skip FIR upsample: upfirdn(x, 4*f2d, up=2, pad=(2,1)) -> NHWC f16 [b][64][64][c]
__global__ void k_skipup(const float* __restrict__ x, half_t* __restrict__ xs)
{
  long idx = (long)blockIdx.x * blockDim.x + threadIdx.x;   // 16*64*64*512
  int c = (int)(idx & 511), xx = (int)((idx >> 9) & 63);
  int y = (int)((idx >> 15) & 63), b = (int)(idx >> 21);
  const float f[4] = {0.125f, 0.375f, 0.375f, 0.125f};
  const float* plane = x + ((long)(b * 512 + c)) * 1024;
  float acc = 0.f;
#pragma unroll
  for (int fy = 0; fy < 4; ++fy) {
    int p = y + fy - 2; if (p & 1) continue;
    int iy = p >> 1;    if (iy < 0 || iy >= 32) continue;
#pragma unroll
    for (int fx = 0; fx < 4; ++fx) {
      int q = xx + fx - 2; if (q & 1) continue;
      int ix = q >> 1;     if (ix < 0 || ix >= 32) continue;
      acc += 4.f * f[fy] * f[fx] * plane[iy * 32 + ix];
    }
  }
  xs[idx] = (half_t)acc;
}

// ---------------------------------------------------------------------------
// conv0: modulated transposed 3x3, stride 2, decomposed by output parity phase
// out(2m+py,2n+px) = sum_taps w0[o,i,2-ky,2-kx] * xm0[i, m-1+ky/2, n-1+kx/2]
// (wT0 holds flipped taps). Wave computes 64(M) x {2 rows} x 16(N); the 4
// A-fragments are shared across both rows: 12 b128 loads per 8 wmma.
// Reads past row/plane ends hit wrapped/slack memory and only land in
// accumulators whose outputs are discarded by the epilogue guards.
// ---------------------------------------------------------------------------
__global__ void __launch_bounds__(256)
k_conv0(const half_t* __restrict__ wT, const half_t* __restrict__ xm,
        half_t* __restrict__ h65, int py, int px)
{
  unsigned wave = threadIdx.x >> 5, lane = threadIdx.x & 31u;
  unsigned g = lane >> 4, mn = lane & 15u;
  int rowsP = 33 - py, colsP = 33 - px;
  int ntX = (colsP + 15) >> 4;
  int mrp = (rowsP + 1) >> 1;                     // row pairs
  long tile = (long)blockIdx.x * 8 + wave;
  int nt = (int)(tile % ntX); tile /= ntX;
  int mp = (int)(tile % mrp); tile /= mrp;
  int bb = (int)(tile % 16);
  int mt = (int)(tile / 16);                      // 0..7: 64 output channels
  int m0 = mp * 2;

  v8f c00 = {}, c01 = {}, c10 = {}, c11 = {}, c20 = {}, c21 = {}, c30 = {}, c31 = {};
  int nky = py ? 1 : 2, nkx = px ? 1 : 2;
  for (int ia = 0; ia < nky; ++ia) {
    int ky  = py ? 1 : ia * 2;
    int iyp = m0 + (py ? 1 : (ky >> 1));          // padded row for m0
    for (int ib = 0; ib < nkx; ++ib) {
      int kx  = px ? 1 : ib * 2;
      int ixp = nt * 16 + (px ? 1 : (kx >> 1));   // padded col base
      int t = ky * 3 + kx;
      const half_t* aBase = wT + ((long)(t * 512 + mt * 64)) * 512;
      const half_t* bBase = xm + (((long)(bb * 34 + iyp)) * 34 + ixp) * 512;
      for (int kb = 0; kb < CIN_; kb += 32) {
        Frag fa0, fa1, fa2, fa3, fb0, fb1;
        load_frag_a(fa0, aBase + kb,            g, mn);
        load_frag_a(fa1, aBase + 16 * 512 + kb, g, mn);
        load_frag_a(fa2, aBase + 32 * 512 + kb, g, mn);
        load_frag_a(fa3, aBase + 48 * 512 + kb, g, mn);
        load_frag_b(fb0, bBase + kb,            g, mn);            // row m0
        load_frag_b(fb1, bBase + 34 * 512 + kb, g, mn);            // row m0+1
        c00 = wmma_step(fa0, fb0, c00);  c01 = wmma_step(fa0, fb1, c01);
        c10 = wmma_step(fa1, fb0, c10);  c11 = wmma_step(fa1, fb1, c11);
        c20 = wmma_step(fa2, fb0, c20);  c21 = wmma_step(fa2, fb1, c21);
        c30 = wmma_step(fa3, fb0, c30);  c31 = wmma_step(fa3, fb1, c31);
      }
    }
  }
  int n = nt * 16 + (int)mn;
  if (n < colsP) {
    v8f* rowAcc0[4] = { &c00, &c10, &c20, &c30 };
    v8f* rowAcc1[4] = { &c01, &c11, &c21, &c31 };
#pragma unroll
    for (int ri = 0; ri < 2; ++ri) {
      int m = m0 + ri;
      if (m < rowsP) {
        int oy = 2 * m + py, ox = 2 * n + px;
        half_t* dst = h65 + ((((long)(bb * 65 + oy)) * 65 + ox) * 512)
                      + mt * 64 + 8 * (int)g;
#pragma unroll
        for (int mi = 0; mi < 4; ++mi) {
          v8f a = ri ? *rowAcc1[mi] : *rowAcc0[mi];
          Pack8 pk;
#pragma unroll
          for (int r = 0; r < 8; ++r) pk.h[r] = (half_t)a[r];
          *reinterpret_cast<uint4*>(dst + mi * 16) = pk.q;   // o = mt*64+mi*16+8g+r
        }
      }
    }
  }
}

// conv0 epilogue: 4x4 FIR (pad 1) on 65x65 -> 64x64, demod, noise, bias,
// relu*sqrt2, pre-modulate by s1, write zero-padded NHWC f16 for conv1.
__global__ void k_fir(const half_t* __restrict__ h65, const float* __restrict__ d0,
                      const float* __restrict__ nc0, const float* __restrict__ ns0,
                      const float* __restrict__ b0, const float* __restrict__ s1,
                      half_t* __restrict__ xm1)
{
  long idx = (long)blockIdx.x * blockDim.x + threadIdx.x;   // 16*64*64*512
  int o = (int)(idx & 511), x = (int)((idx >> 9) & 63);
  int y = (int)((idx >> 15) & 63), b = (int)(idx >> 21);
  const float f[4] = {0.125f, 0.375f, 0.375f, 0.125f};
  float acc = 0.f;
#pragma unroll
  for (int fy = 0; fy < 4; ++fy) {
    int iy = y + fy - 1; if (iy < 0 || iy >= 65) continue;
#pragma unroll
    for (int fx = 0; fx < 4; ++fx) {
      int ix = x + fx - 1; if (ix < 0 || ix >= 65) continue;
      acc += 4.f * f[fy] * f[fx]
             * (float)h65[(((long)(b * 65 + iy)) * 65 + ix) * 512 + o];
    }
  }
  int bo = b * 512 + o;
  float v = acc * d0[bo] + nc0[y * 64 + x] * ns0[0] + b0[o];
  v = fmaxf(v, 0.f) * SQ2_;
  xm1[(((long)(b * 66 + (y + 1))) * 66 + (x + 1)) * 512 + o] = (half_t)(v * s1[bo]);
}

// skip 1x1 conv: GEMM M=512, N=B*64*64, K=512; wave computes 64x32; writes d_out
__global__ void __launch_bounds__(256)
k_skipgemm(const half_t* __restrict__ wsk, const half_t* __restrict__ xsh,
           float* __restrict__ out)
{
  unsigned wave = threadIdx.x >> 5, lane = threadIdx.x & 31u;
  unsigned g = lane >> 4, mn = lane & 15u;
  long tile = (long)blockIdx.x * 8 + wave;     // 8(mt)*16(b)*64(y)*2(xt)
  int xt = (int)(tile & 1), y = (int)((tile >> 1) & 63);
  int bb = (int)((tile >> 7) & 15), mt = (int)(tile >> 11);
  v8f c00 = {}, c01 = {}, c10 = {}, c11 = {}, c20 = {}, c21 = {}, c30 = {}, c31 = {};
  const half_t* aBase = wsk + (long)(mt * 64) * 512;
  const half_t* bBase = xsh + (((long)(bb * 64 + y)) * 64 + xt * 32) * 512;
  for (int kb = 0; kb < CIN_; kb += 32) {
    Frag fa0, fa1, fa2, fa3, fb0, fb1;
    load_frag_a(fa0, aBase + kb,            g, mn);
    load_frag_a(fa1, aBase + 16 * 512 + kb, g, mn);
    load_frag_a(fa2, aBase + 32 * 512 + kb, g, mn);
    load_frag_a(fa3, aBase + 48 * 512 + kb, g, mn);
    load_frag_b(fb0, bBase + kb,            g, mn);
    load_frag_b(fb1, bBase + 16 * 512 + kb, g, mn);
    c00 = wmma_step(fa0, fb0, c00);  c01 = wmma_step(fa0, fb1, c01);
    c10 = wmma_step(fa1, fb0, c10);  c11 = wmma_step(fa1, fb1, c11);
    c20 = wmma_step(fa2, fb0, c20);  c21 = wmma_step(fa2, fb1, c21);
    c30 = wmma_step(fa3, fb0, c30);  c31 = wmma_step(fa3, fb1, c31);
  }
  int x0 = xt * 32 + (int)mn, x1 = x0 + 16;
  v8f* acc0[4] = { &c00, &c10, &c20, &c30 };
  v8f* acc1[4] = { &c01, &c11, &c21, &c31 };
#pragma unroll
  for (int mi = 0; mi < 4; ++mi) {
#pragma unroll
    for (int r = 0; r < 8; ++r) {
      int o = mt * 64 + mi * 16 + r + 8 * (int)g;
      long base = (((long)(bb * 512 + o)) * 64 + y) * 64;
      out[base + x0] = (*acc0[mi])[r] * SQH_;
      out[base + x1] = (*acc1[mi])[r] * SQH_;
    }
  }
}

// conv1: 3x3 pad 1 implicit GEMM on padded NHWC xm1; wave computes 64x32;
// fused demod/noise/bias/relu; accumulates onto d_out (holds skip branch).
__global__ void __launch_bounds__(256)
k_conv1(const half_t* __restrict__ wT, const half_t* __restrict__ xm1,
        const float* __restrict__ d1, const float* __restrict__ nc1,
        const float* __restrict__ ns1, const float* __restrict__ b1,
        float* __restrict__ out)
{
  unsigned wave = threadIdx.x >> 5, lane = threadIdx.x & 31u;
  unsigned g = lane >> 4, mn = lane & 15u;
  long tile = (long)blockIdx.x * 8 + wave;     // 8(mt)*16(b)*64(y)*2(xt)
  int xt = (int)(tile & 1), y = (int)((tile >> 1) & 63);
  int bb = (int)((tile >> 7) & 15), mt = (int)(tile >> 11);
  v8f c00 = {}, c01 = {}, c10 = {}, c11 = {}, c20 = {}, c21 = {}, c30 = {}, c31 = {};
  for (int t = 0; t < 9; ++t) {
    int ky = t / 3, kx = t - 3 * ky;
    const half_t* aBase = wT + ((long)(t * 512 + mt * 64)) * 512;
    const half_t* bBase = xm1 + (((long)(bb * 66 + y + ky)) * 66 + xt * 32 + kx) * 512;
    for (int kb = 0; kb < COUT_; kb += 32) {
      Frag fa0, fa1, fa2, fa3, fb0, fb1;
      load_frag_a(fa0, aBase + kb,            g, mn);
      load_frag_a(fa1, aBase + 16 * 512 + kb, g, mn);
      load_frag_a(fa2, aBase + 32 * 512 + kb, g, mn);
      load_frag_a(fa3, aBase + 48 * 512 + kb, g, mn);
      load_frag_b(fb0, bBase + kb,            g, mn);
      load_frag_b(fb1, bBase + 16 * 512 + kb, g, mn);
      c00 = wmma_step(fa0, fb0, c00);  c01 = wmma_step(fa0, fb1, c01);
      c10 = wmma_step(fa1, fb0, c10);  c11 = wmma_step(fa1, fb1, c11);
      c20 = wmma_step(fa2, fb0, c20);  c21 = wmma_step(fa2, fb1, c21);
      c30 = wmma_step(fa3, fb0, c30);  c31 = wmma_step(fa3, fb1, c31);
    }
  }
  int x0 = xt * 32 + (int)mn, x1 = x0 + 16;
  float nsv = ns1[0];
  float nz0 = nc1[y * 64 + x0] * nsv, nz1 = nc1[y * 64 + x1] * nsv;
  v8f* acc0[4] = { &c00, &c10, &c20, &c30 };
  v8f* acc1[4] = { &c01, &c11, &c21, &c31 };
#pragma unroll
  for (int mi = 0; mi < 4; ++mi) {
#pragma unroll
    for (int r = 0; r < 8; ++r) {
      int o = mt * 64 + mi * 16 + r + 8 * (int)g;
      float dd = d1[bb * 512 + o], bv = b1[o];
      long base = (((long)(bb * 512 + o)) * 64 + y) * 64;
      float v0 = fmaxf((*acc0[mi])[r] * dd + nz0 + bv, 0.f) * (SQ2_ * SQH_);
      float v1 = fmaxf((*acc1[mi])[r] * dd + nz1 + bv, 0.f) * (SQ2_ * SQH_);
      out[base + x0] += v0;
      out[base + x1] += v1;
    }
  }
}

// ---------------------------------------------------------------------------
// Host launcher
// ---------------------------------------------------------------------------
extern "C" void kernel_launch(void* const* d_in, const int* in_sizes, int n_in,
                              void* d_out, int out_size, void* d_ws, size_t ws_size,
                              hipStream_t stream)
{
  (void)in_sizes; (void)n_in; (void)out_size; (void)ws_size;
  const float* x       = (const float*)d_in[0];
  // d_in[1] = latent1 (unused by the reference computation)
  const float* latent2 = (const float*)d_in[2];
  const float* aff0_w  = (const float*)d_in[3];
  const float* aff0_b  = (const float*)d_in[4];
  const float* w0      = (const float*)d_in[5];
  const float* b0      = (const float*)d_in[6];
  const float* nc0     = (const float*)d_in[7];
  const float* ns0     = (const float*)d_in[8];
  const float* aff1_w  = (const float*)d_in[9];
  const float* aff1_b  = (const float*)d_in[10];
  const float* w1      = (const float*)d_in[11];
  const float* b1      = (const float*)d_in[12];
  const float* nc1     = (const float*)d_in[13];
  const float* ns1     = (const float*)d_in[14];
  const float* skip_w  = (const float*)d_in[15];
  float* out = (float*)d_out;
  char*  ws  = (char*)d_ws;

  // workspace layout (256B-aligned sizes; 64KB slack after xm0 for B overrun)
  float*  s0    = (float*) (ws + 0);          //  32768
  float*  s1    = (float*) (ws + 32768);      //  32768
  float*  dm0   = (float*) (ws + 65536);      //  32768
  float*  dm1   = (float*) (ws + 98304);      //  32768
  float*  wsum0 = (float*) (ws + 131072);     //  1 MB
  float*  wsum1 = (float*) (ws + 1179648);    //  1 MB
  half_t* wT0   = (half_t*)(ws + 2228224);    //  9*512*512 f16
  half_t* wT1   = (half_t*)(ws + 6946816);    //  9*512*512 f16
  half_t* wsk   = (half_t*)(ws + 11665408);   //  512*512 f16
  half_t* xm0   = (half_t*)(ws + 12189696);   //  16*34*34*512 f16 (+64KB slack)
  half_t* xsh   = (half_t*)(ws + 31195136);   //  16*64*64*512 f16
  half_t* h65   = (half_t*)(ws + 98304000);   //  16*65*65*512 f16
  half_t* xm1   = (half_t*)(ws + 167526400);  //  16*66*66*512 f16 (padded)

  // styles, demod coefficients, weight conversion
  k_affine<<<32, 256, 0, stream>>>(latent2 + 0,   1024, aff0_w, aff0_b, s0);
  k_affine<<<32, 256, 0, stream>>>(latent2 + 512, 1024, aff1_w, aff1_b, s1);
  k_wsum  <<<1024, 256, 0, stream>>>(w0, wsum0);
  k_wsum  <<<1024, 256, 0, stream>>>(w1, wsum1);
  k_demod <<<32, 256, 0, stream>>>(s0, wsum0, dm0);
  k_demod <<<32, 256, 0, stream>>>(s1, wsum1, dm1);
  k_wprep <<<9216, 256, 0, stream>>>(w0, wT0, 1);   // flipped (conv_transpose)
  k_wprep <<<9216, 256, 0, stream>>>(w1, wT1, 0);
  k_wskp  <<<1024, 256, 0, stream>>>(skip_w, wsk);

  // activation prep (NHWC f16)
  k_xmod0 <<<36992,  256, 0, stream>>>(x, s0, xm0);
  k_skipup<<<131072, 256, 0, stream>>>(x, xsh);

  // conv0: four parity phases of the stride-2 transposed conv (WMMA)
  for (int py = 0; py < 2; ++py) {
    for (int px = 0; px < 2; ++px) {
      int rowsP = 33 - py;
      int mrp   = (rowsP + 1) >> 1;
      int ntX   = ((33 - px) + 15) >> 4;
      int blocks = 16 * mrp * ntX;   // 8(mt)*16(b)*mrp*ntX tiles / 8 waves
      k_conv0<<<blocks, 256, 0, stream>>>(wT0, xm0, h65, py, px);
    }
  }

  // FIR + demod + noise + bias + relu, pre-modulated for conv1 (padded NHWC)
  (void)hipMemsetAsync(xm1, 0, 71368704, stream);   // zero padded border
  k_fir<<<131072, 256, 0, stream>>>(h65, dm0, nc0, ns0, b0, s1, xm1);

  // skip branch GEMM writes d_out, conv1 GEMM accumulates on top
  k_skipgemm<<<2048, 256, 0, stream>>>(wsk, xsh, out);
  k_conv1   <<<2048, 256, 0, stream>>>(wT1, xm1, dm1, nc1, ns1, b1, out);
}
